// LiquidFusion_30356828848373
// MI455X (gfx1250) — compile-verified, matrix operand order checked by reading, not run on hardware
//
#include <hip/hip_runtime.h>
#include <hip/hip_bf16.h>
#include <math.h>

// ---------------------------------------------------------------------------
// MI455X / gfx1250 fused attention: bf16 WMMA (v_wmma_f32_16x16x32_bf16),
// flash-attention with TDM (tensor_load_to_lds) double-buffered K/V staging,
// exp2-domain streaming softmax, WMMA-based row sums, diagonal-peeled mask.
// ---------------------------------------------------------------------------

typedef __bf16 bf16_t;
typedef __attribute__((ext_vector_type(16))) __bf16 v16bf;
typedef __attribute__((ext_vector_type(8)))  __bf16 v8bf;
typedef __attribute__((ext_vector_type(8)))  float  v8f;
typedef __attribute__((ext_vector_type(4)))  unsigned int u32x4;
typedef __attribute__((ext_vector_type(4)))  int i32x4;
typedef __attribute__((ext_vector_type(8)))  int i32x8;

#define QLEN   8192
#define HID    1024
#define NHEADS 16
#define HDIM   64
// softmax scale folded into Q at RoPE time: 1/sqrt(64) * log2(e)
#define QSCALE (0.125f * 1.44269504088896f)

#if defined(__has_builtin)
#if __has_builtin(__builtin_amdgcn_tensor_load_to_lds) && \
    __has_builtin(__builtin_amdgcn_s_wait_tensorcnt)
#define USE_TDM 1
#endif
#endif
#ifndef USE_TDM
#define USE_TDM 0
#endif

// --------------------------------------------------------------------------
// Fragment loader for 16-bit WMMA operands (A 16x32 M-major / B 32x16 K-major,
// both fed from row-major storage where each "row" supplies one lane):
//   lanes 0-15 : row = lane,    k-chunks [0,8) and [16,24)
//   lanes 16-31: row = lane-16, k-chunks [8,16) and [24,32)
// Two 16-byte contiguous loads per lane (global_load_b128 / ds_load_b128).
// --------------------------------------------------------------------------
__device__ __forceinline__ v16bf load_frag(const bf16_t* p, int ld, int lane) {
  const int row = lane & 15;
  const int hi  = lane >> 4;
  const bf16_t* r = p + (size_t)row * ld + hi * 8;
  v8bf c0 = *(const v8bf*)(r);
  v8bf c1 = *(const v8bf*)(r + 16);
  v16bf out;
#pragma unroll
  for (int i = 0; i < 8; ++i) { out[i] = c0[i]; out[8 + i] = c1[i]; }
  return out;
}

__device__ __forceinline__ v8f wmma32(v16bf a, v16bf b, v8f c) {
  // (neg_a, A, neg_b, B, c_mod, C, reuse_a, reuse_b)
  return __builtin_amdgcn_wmma_f32_16x16x32_bf16(false, a, false, b,
                                                 (short)0, c, false, false);
}

// --------------------------------------------------------------------------
// TDM 2D tile load: tile_w contiguous elements per row, tile_h rows with a
// row stride of stride_elems, bf16 data, packed into LDS at lds_dst.
// D# layout per CDNA5 ISA ch.8 (group0: count/lds_addr/global_addr/type=2;
// group1: data_size=1(2B), tensor_dim0/1 = tile dims (no OOB), tile_dim0/1,
// tensor_dim0_stride).  Groups 2/3 zero (2D tile, tile_dim2/3 unused).
// --------------------------------------------------------------------------
#if USE_TDM
__device__ __forceinline__ void tdm_load_tile_2d(void* lds_dst, const bf16_t* src,
                                                 unsigned tile_w, unsigned tile_h,
                                                 unsigned stride_elems) {
  const unsigned long long ga = (unsigned long long)(uintptr_t)src;
  const unsigned lds_off = (unsigned)(uintptr_t)lds_dst; // low 32b = LDS byte addr
  u32x4 g0;
  g0[0] = 1u;                                            // count=1, user D#
  g0[1] = lds_off;                                       // lds_addr
  g0[2] = (unsigned)ga;                                  // global_addr[31:0]
  g0[3] = (unsigned)((ga >> 32) & 0x01FFFFFFu) | (2u << 30); // addr[56:32]|type=2
  i32x8 g1;
  g1[0] = 0x00010000;                                    // data_size=1 (2 bytes)
  g1[1] = (int)((tile_w & 0xFFFFu) << 16);               // tensor_dim0[15:0]
  g1[2] = (int)((tile_w >> 16) | ((tile_h & 0xFFFFu) << 16)); // d0 hi | tensor_dim1 lo
  g1[3] = (int)((tile_h >> 16) | (tile_w << 16));        // d1 hi | tile_dim0
  g1[4] = (int)tile_h;                                   // tile_dim1 (tile_dim2=0)
  g1[5] = (int)stride_elems;                             // tensor_dim0_stride[31:0]
  g1[6] = 0;
  g1[7] = 0;
  const i32x4 gz = {0, 0, 0, 0};
#if __clang_major__ >= 23
  const i32x8 gz8 = {0, 0, 0, 0, 0, 0, 0, 0};
  __builtin_amdgcn_tensor_load_to_lds(g0, g1, gz, gz, gz8, 0);
#else
  __builtin_amdgcn_tensor_load_to_lds(g0, g1, gz, gz, 0);
#endif
}
#endif

// --------------------------------------------------------------------------
// fp32 -> bf16 conversion
// --------------------------------------------------------------------------
__global__ void f32_to_bf16(const float* __restrict__ src,
                            bf16_t* __restrict__ dst, int n) {
  int i = blockIdx.x * blockDim.x + threadIdx.x;
  int stride = gridDim.x * blockDim.x;
  for (; i < n; i += stride) dst[i] = (bf16_t)src[i];
}

// --------------------------------------------------------------------------
// Generic C = A * B^T GEMM (A: MxK row-major bf16, B: NxK row-major bf16).
// Block = 128 threads (4 waves), block tile 128x64; each wave owns a 32-row
// strip (2 row-tiles x 4 col-tiles = 8 WMMA per k-step).  OUT_MODE: 0 = bf16
// row-major, 1 = bf16 transposed store (produces Vt), 2 = fp32 row-major.
// --------------------------------------------------------------------------
template <int OUT_MODE>
__global__ __launch_bounds__(128)
void gemm_bt(const bf16_t* __restrict__ A, const bf16_t* __restrict__ B,
             void* __restrict__ C, int M, int N, int K) {
  const int lane = threadIdx.x & 31;
  const int w    = threadIdx.x >> 5;
  const int bn   = blockIdx.x, bm = blockIdx.y;
  const int row0 = bm * 128 + w * 32;
  v8f acc[2][4] = {};
  for (int k0 = 0; k0 < K; k0 += 32) {
    v16bf a0 = load_frag(A + (size_t)row0 * K + k0,        K, lane);
    v16bf a1 = load_frag(A + (size_t)(row0 + 16) * K + k0, K, lane);
#pragma unroll
    for (int t = 0; t < 4; ++t) {
      v16bf b = load_frag(B + (size_t)(bn * 64 + t * 16) * K + k0, K, lane);
      acc[0][t] = wmma32(a0, b, acc[0][t]);
      acc[1][t] = wmma32(a1, b, acc[1][t]);
    }
  }
  const int hi = lane >> 4, nn = lane & 15;
#pragma unroll
  for (int g = 0; g < 2; ++g)
#pragma unroll
    for (int t = 0; t < 4; ++t) {
      const int col = bn * 64 + t * 16 + nn;
#pragma unroll
      for (int r = 0; r < 8; ++r) {
        const int row = row0 + g * 16 + r + 8 * hi;
        const float v = acc[g][t][r];
        if (OUT_MODE == 0)      ((bf16_t*)C)[(size_t)row * N + col] = (bf16_t)v;
        else if (OUT_MODE == 1) ((bf16_t*)C)[(size_t)col * M + row] = (bf16_t)v;
        else                    ((float*)C)[(size_t)row * N + col]  = v;
      }
    }
}

// --------------------------------------------------------------------------
// RoPE in place on bf16 Q and K; softmax scale folded into Q (exp2 domain).
// --------------------------------------------------------------------------
__global__ void rope_kernel(bf16_t* __restrict__ Qm, bf16_t* __restrict__ Km) {
  const int idx = blockIdx.x * blockDim.x + threadIdx.x;
  if (idx >= QLEN * NHEADS * 32) return;
  const int d = idx & 31;
  const int h = (idx >> 5) & (NHEADS - 1);
  const int q = idx >> 9;
  const float inv = __expf(-logf(10000.0f) * (float)d / 32.0f); // 10000^(-2d/64)
  float sv, cv;
  __sincosf((float)q * inv, &sv, &cv);
  const size_t i1 = (size_t)q * HID + h * HDIM + d;
  const size_t i2 = i1 + 32;
  const float q1 = (float)Qm[i1], q2 = (float)Qm[i2];
  Qm[i1] = (bf16_t)((q1 * cv - q2 * sv) * QSCALE);
  Qm[i2] = (bf16_t)((q2 * cv + q1 * sv) * QSCALE);
  const float k1 = (float)Km[i1], k2 = (float)Km[i2];
  Km[i1] = (bf16_t)(k1 * cv - k2 * sv);
  Km[i2] = (bf16_t)(k2 * cv + k1 * sv);
}

// --------------------------------------------------------------------------
// One 128-key attention step for a wave's 16 query rows; K/V tiles come from
// LDS (Kt: 128 keys x 64 dims, ld 64; Vv: 64 dims x 128 keys, ld 128).
// --------------------------------------------------------------------------
template <int NT, bool MASK>
__device__ __forceinline__ void attn_block(
    const bf16_t* Kt, const bf16_t* Vv, int kk, int q0, int lane,
    const v16bf& qa0, const v16bf& qa1, const v16bf& onesB,
    bf16_t* myP, v8f* o, float* mx, float* l) {
  const int hi = lane >> 4, nn = lane & 15;
  // ---- S tiles: 2 chained WMMAs each over the 64-dim head ----
  v8f s[NT];
#pragma unroll
  for (int t = 0; t < NT; ++t) {
    v16bf b0 = load_frag(Kt + (t * 16) * 64,      64, lane);
    v16bf b1 = load_frag(Kt + (t * 16) * 64 + 32, 64, lane);
    v8f z = {};
    z    = wmma32(qa0, b0, z);
    s[t] = wmma32(qa1, b1, z);
  }
  if (MASK) {
#pragma unroll
    for (int t = 0; t < NT; ++t)
#pragma unroll
      for (int r = 0; r < 8; ++r) {
        const int qrow = q0 + r + 8 * hi;
        const int key  = kk + t * 16 + nn;
        s[t][r] = (key > qrow) ? -1e30f : s[t][r];
      }
  }
  // ---- online softmax (max via half-wave shuffles; exp2 native) ----
  float alpha[8];
#pragma unroll
  for (int r = 0; r < 8; ++r) {
    float smax = s[0][r];
#pragma unroll
    for (int t = 1; t < NT; ++t) smax = fmaxf(smax, s[t][r]);
    for (int off = 1; off < 16; off <<= 1)
      smax = fmaxf(smax, __shfl_xor(smax, off, 32));
    const float mnew = fmaxf(mx[r], smax);
    alpha[r] = exp2f(mx[r] - mnew);
    mx[r] = mnew;
#pragma unroll
    for (int t = 0; t < NT; ++t) {
      const float p = exp2f(s[t][r] - mnew);
      myP[(r + 8 * hi) * 128 + t * 16 + nn] = (bf16_t)p;
    }
  }
  // ---- P A-fragments from this wave's private LDS; row sums via ones-WMMA --
  v16bf pa[NT / 2];
#pragma unroll
  for (int j = 0; j < NT / 2; ++j)
    pa[j] = load_frag(myP + j * 32, 128, lane);
  v8f rs = {};
#pragma unroll
  for (int j = 0; j < NT / 2; ++j) rs = wmma32(pa[j], onesB, rs);
#pragma unroll
  for (int r = 0; r < 8; ++r) {
    l[r] = l[r] * alpha[r] + rs[r];
#pragma unroll
    for (int t = 0; t < 4; ++t) o[t][r] *= alpha[r];
  }
  // ---- O += P * V ----
#pragma unroll
  for (int t = 0; t < 4; ++t)
#pragma unroll
    for (int j = 0; j < NT / 2; ++j) {
      v16bf vb = load_frag(Vv + (t * 16) * 128 + j * 32, 128, lane);
      o[t] = wmma32(pa[j], vb, o[t]);
    }
}

// --------------------------------------------------------------------------
// Flash attention.  Grid = (QLEN/128, NHEADS).  Block = 256 threads (8
// waves); wave w owns 16 query rows.  K/V 128-key tiles are staged into LDS
// by the Tensor Data Mover (wave 0 issues tensor_load_to_lds, waits on
// TENSORcnt, barrier publishes), double buffered so block i+1's DMA overlaps
// block i's 36 WMMAs.  Diagonal block peeled into the masked instantiation.
// --------------------------------------------------------------------------
__global__ __launch_bounds__(256)
void flash_attn(const bf16_t* __restrict__ Qm, const bf16_t* __restrict__ Km,
                const bf16_t* __restrict__ Vt, bf16_t* __restrict__ Om) {
  __shared__ __attribute__((aligned(16))) bf16_t Pl[8 * 16 * 128];   // 32 KB
  __shared__ __attribute__((aligned(16))) bf16_t Kl[2][128 * 64];    // 2x16 KB
  __shared__ __attribute__((aligned(16))) bf16_t Vl[2][64 * 128];    // 2x16 KB
  const int lane = threadIdx.x & 31;
  const int w    = threadIdx.x >> 5;   // 0..7
  const int qb   = blockIdx.x;         // 0..63
  const int h    = blockIdx.y;
  const int q0   = qb * 128 + w * 16;
  const int hi = lane >> 4, nn = lane & 15;

  const v16bf qa0 = load_frag(Qm + (size_t)q0 * HID + h * HDIM,      HID, lane);
  const v16bf qa1 = load_frag(Qm + (size_t)q0 * HID + h * HDIM + 32, HID, lane);
  v16bf onesB;
#pragma unroll
  for (int i = 0; i < 16; ++i) onesB[i] = (bf16_t)1.0f;

  v8f o[4] = {};
  float mx[8], l[8];
#pragma unroll
  for (int r = 0; r < 8; ++r) { mx[r] = -1e30f; l[r] = 0.0f; }
  bf16_t* myP = Pl + w * (16 * 128);

#if USE_TDM
  // ---- prologue: DMA block 0 into buffer 0 ----
  if (w == 0) {
    tdm_load_tile_2d(Kl[0], Km + (size_t)0 * HID + h * HDIM, HDIM, 128, HID);
    tdm_load_tile_2d(Vl[0], Vt + (size_t)(h * HDIM) * QLEN + 0, 128, HDIM, QLEN);
  }
  for (int i = 0; i < qb; ++i) {       // uniform trip count
    const int kk = i * 128;
    if (w == 0) {
      // issue block i+1 into the other buffer, then wait for block i's pair
      tdm_load_tile_2d(Kl[(i + 1) & 1], Km + (size_t)(kk + 128) * HID + h * HDIM,
                       HDIM, 128, HID);
      tdm_load_tile_2d(Vl[(i + 1) & 1], Vt + (size_t)(h * HDIM) * QLEN + kk + 128,
                       128, HDIM, QLEN);
      __builtin_amdgcn_s_wait_tensorcnt(2);
    }
    __syncthreads();                   // publish buffer i&1 to all waves
    attn_block<8, false>(Kl[i & 1], Vl[i & 1], kk, q0, lane,
                         qa0, qa1, onesB, myP, o, mx, l);
    __syncthreads();                   // buffer i&1 free for overwrite
  }
  if (w == 0) __builtin_amdgcn_s_wait_tensorcnt(0);
  __syncthreads();
  attn_block<8, true>(Kl[qb & 1], Vl[qb & 1], qb * 128, q0, lane,
                      qa0, qa1, onesB, myP, o, mx, l);
#else
  // ---- fallback: cooperative global->LDS staging (single buffer) ----
  for (int i = 0; i <= qb; ++i) {
    const int kk = i * 128;
    const int tid = threadIdx.x;
    __syncthreads();
    for (int c = tid; c < 128 * 8; c += 256) {      // K tile: 8-elem chunks
      const int row = c >> 3, cc = (c & 7) * 8;
      *(v8bf*)(Kl[0] + row * 64 + cc) =
          *(const v8bf*)(Km + (size_t)(kk + row) * HID + h * HDIM + cc);
    }
    for (int c = tid; c < 64 * 16; c += 256) {      // V tile: 8-elem chunks
      const int row = c >> 4, cc = (c & 15) * 8;
      *(v8bf*)(Vl[0] + row * 128 + cc) =
          *(const v8bf*)(Vt + (size_t)(h * HDIM + row) * QLEN + kk + cc);
    }
    __syncthreads();
    if (i < qb)
      attn_block<8, false>(Kl[0], Vl[0], kk, q0, lane,
                           qa0, qa1, onesB, myP, o, mx, l);
    else
      attn_block<8, true>(Kl[0], Vl[0], kk, q0, lane,
                          qa0, qa1, onesB, myP, o, mx, l);
  }
#endif

  // ---- epilogue: normalize and store bf16 attention output ----
#pragma unroll
  for (int r = 0; r < 8; ++r) {
    const float inv = 1.0f / l[r];
    const int row = q0 + r + 8 * hi;
#pragma unroll
    for (int t = 0; t < 4; ++t)
      Om[(size_t)row * HID + h * HDIM + t * 16 + nn] = (bf16_t)(o[t][r] * inv);
  }
}

// --------------------------------------------------------------------------
// Host-side orchestration (all on `stream`, workspace only, graph-safe).
// Inputs: 0 hidden_states, 1 attention_mask (unused: mask is analytic),
//         2 position_ids (unused: positions are arange), 3..6 Wq,Wk,Wv,Wo.
// --------------------------------------------------------------------------
extern "C" void kernel_launch(void* const* d_in, const int* in_sizes, int n_in,
                              void* d_out, int out_size, void* d_ws, size_t ws_size,
                              hipStream_t stream) {
  (void)in_sizes; (void)n_in; (void)out_size; (void)ws_size;
  const float* X  = (const float*)d_in[0];
  const float* Wq = (const float*)d_in[3];
  const float* Wk = (const float*)d_in[4];
  const float* Wv = (const float*)d_in[5];
  const float* Wo = (const float*)d_in[6];

  const size_t XN = (size_t)QLEN * HID;  // 8,388,608
  const size_t WN = (size_t)HID * HID;   // 1,048,576

  bf16_t* p   = (bf16_t*)d_ws;
  bf16_t* Xb  = p; p += XN;
  bf16_t* Wqb = p; p += WN;
  bf16_t* Wkb = p; p += WN;
  bf16_t* Wvb = p; p += WN;
  bf16_t* Wob = p; p += WN;
  bf16_t* Qb  = p; p += XN;
  bf16_t* Kb  = p; p += XN;
  bf16_t* Vtb = p; p += XN;   // V^T per head: [h][d][k], k contiguous
  bf16_t* Ob  = p; p += XN;   // attention output (pre-Wo), bf16

  f32_to_bf16<<<4096, 256, 0, stream>>>(X,  Xb,  (int)XN);
  f32_to_bf16<<<1024, 256, 0, stream>>>(Wq, Wqb, (int)WN);
  f32_to_bf16<<<1024, 256, 0, stream>>>(Wk, Wkb, (int)WN);
  f32_to_bf16<<<1024, 256, 0, stream>>>(Wv, Wvb, (int)WN);
  f32_to_bf16<<<1024, 256, 0, stream>>>(Wo, Wob, (int)WN);

  const dim3 gg(HID / 64, QLEN / 128);
  gemm_bt<0><<<gg, 128, 0, stream>>>(Xb, Wqb, Qb,  QLEN, HID, HID);
  gemm_bt<0><<<gg, 128, 0, stream>>>(Xb, Wkb, Kb,  QLEN, HID, HID);
  gemm_bt<1><<<gg, 128, 0, stream>>>(Xb, Wvb, Vtb, QLEN, HID, HID);  // store V^T

  rope_kernel<<<(QLEN * NHEADS * 32) / 256, 256, 0, stream>>>(Qb, Kb);

  flash_attn<<<dim3(QLEN / 128, NHEADS), 256, 0, stream>>>(Qb, Kb, Vtb, Ob);

  gemm_bt<2><<<gg, 128, 0, stream>>>(Ob, Wob, d_out, QLEN, HID, HID);  // fp32 out
}
